// Modified_SAGE_19301583029054
// MI455X (gfx1250) — matrix operand order repaired and checked by reading, not compiled, not run on hardware
//
#include <hip/hip_runtime.h>
#include <hip/hip_bf16.h>

typedef __attribute__((ext_vector_type(16))) __bf16 v16bf;
typedef __attribute__((ext_vector_type(8)))  __bf16 v8bf;
typedef __attribute__((ext_vector_type(8)))  float  v8f;

#define DH 128
#define KTOT 256  // concatenated [agg | x] K dimension

// ---------------- utility kernels ----------------

__global__ void k_zero(float* __restrict__ p, long long n) {
    long long i = (long long)blockIdx.x * blockDim.x + threadIdx.x;
    if (i < n) p[i] = 0.0f;
}

__global__ void k_deg(const int* __restrict__ dst, float* __restrict__ deg, int E) {
    int e = blockIdx.x * blockDim.x + threadIdx.x;
    if (e < E) atomicAdd(&deg[dst[e]], 1.0f);
}

__global__ void k_invdeg(float* __restrict__ deg, int n) {
    int i = blockIdx.x * blockDim.x + threadIdx.x;
    if (i < n) deg[i] = 1.0f / fmaxf(deg[i], 1.0f);
}

// one thread per (edge, 4-float chunk): coalesced float4 gather + 4 atomic f32 scatter-adds
__global__ void k_scatter(const float* __restrict__ x, const int* __restrict__ src,
                          const int* __restrict__ dst, float* __restrict__ agg, int E) {
    long long tid = (long long)blockIdx.x * blockDim.x + threadIdx.x;
    long long total = (long long)E * (DH / 4);
    if (tid >= total) return;
    int e = (int)(tid >> 5);          // DH/4 == 32 chunks per edge
    int c = ((int)tid & 31) * 4;
    int s = src[e], d = dst[e];
    float4 v = *(const float4*)(x + (size_t)s * DH + c);
    float* p = agg + (size_t)d * DH + c;
    atomicAdd(p + 0, v.x);
    atomicAdd(p + 1, v.y);
    atomicAdd(p + 2, v.z);
    atomicAdd(p + 3, v.w);
}

// pack bf16 A[N][256] = [ agg*inv_deg (0..127) | x (128..255) ]
__global__ void k_prep(const float* __restrict__ x, const float* __restrict__ agg,
                       const float* __restrict__ invdeg, __bf16* __restrict__ Abf, int N) {
    long long tid = (long long)blockIdx.x * blockDim.x + threadIdx.x;
    long long total = (long long)N * (KTOT / 8);
    if (tid >= total) return;
    int n = (int)(tid >> 5);          // KTOT/8 == 32 chunks per row
    int c = ((int)tid & 31) * 8;
    v8bf o;
    if (c < DH) {
        float inv = invdeg[n];
        const float* a = agg + (size_t)n * DH + c;
        #pragma unroll
        for (int j = 0; j < 8; ++j) o[j] = (__bf16)(a[j] * inv);
    } else {
        const float* a = x + (size_t)n * DH + (c - DH);
        #pragma unroll
        for (int j = 0; j < 8; ++j) o[j] = (__bf16)a[j];
    }
    *(v8bf*)(Abf + (size_t)n * KTOT + c) = o;
}

// pack bf16 W[dout][256] = [ Wl row | Wr row ]
__global__ void k_convw(const float* __restrict__ Wl, const float* __restrict__ Wr,
                        __bf16* __restrict__ Wbf, int dout) {
    int tid = blockIdx.x * blockDim.x + threadIdx.x;
    if (tid >= dout * KTOT) return;
    int row = tid >> 8, col = tid & 255;
    float v = (col < DH) ? Wl[row * DH + col] : Wr[row * DH + (col - DH)];
    Wbf[(size_t)row * KTOT + col] = (__bf16)v;
}

// ---------------- WMMA GEMM: out[n, j] = relu( A[n,:] . W[j,:] + bias[j] ) ----------------
// block = 256 threads = 8 waves; each wave owns a 16x16 output tile.
// rowtiles * coltiles == 8;  coltiles = dout/16.
__global__ void k_gemm(const __bf16* __restrict__ A, const __bf16* __restrict__ W,
                       const float* __restrict__ bias, float* __restrict__ out,
                       int nrows, int dout, int rowtiles, int coltiles, int relu) {
    int wave = threadIdx.x >> 5;
    int lane = threadIdx.x & 31;
    int r = wave / coltiles;
    int c = wave % coltiles;
    int base = blockIdx.x * (16 * rowtiles);

    int mrow = base + r * 16 + (lane & 15);
    if (mrow >= nrows) mrow = nrows - 1;          // clamp loads; stores are guarded
    int ncol = c * 16 + (lane & 15);

    // fragment k order used here (wave32; A and B use the same order):
    //  A 16x32 bf16: lane<16 -> K 0..15 consecutive; lane>=16 -> K 16..31
    //  B 32x16 bf16: lane<16 -> K 0..15 consecutive; lane>=16 -> K 16..31
    int khiA = (lane & 16) ? 16 : 0;
    int kbB  = (lane & 16) ? 16 : 0;
    const __bf16* arow = A + (size_t)mrow * KTOT + khiA;
    const __bf16* wrow = W + (size_t)ncol * KTOT + kbB;

    v8f acc = {};
    #pragma unroll
    for (int kk = 0; kk < KTOT; kk += 32) {
        v8bf alo = *(const v8bf*)(arow + kk);
        v8bf ahi = *(const v8bf*)(arow + kk + 8);
        v16bf a = __builtin_shufflevector(alo, ahi, 0, 1, 2, 3, 4, 5, 6, 7,
                                          8, 9, 10, 11, 12, 13, 14, 15);
        v16bf b = *(const v16bf*)(wrow + kk);
        acc = __builtin_amdgcn_wmma_f32_16x16x32_bf16(
            /*neg_a=*/false, a, /*neg_b=*/false, b,
            /*c_mod=*/(short)0, acc, /*reuse_a=*/false, /*reuse_b=*/false);
    }

    // C/D layout: element i -> row (base + r*16 + i + (lane>=16 ? 8 : 0)), col = c*16 + (lane&15)
    float bv = bias[ncol];
    int rowb = base + r * 16 + ((lane & 16) ? 8 : 0);
    #pragma unroll
    for (int i = 0; i < 8; ++i) {
        int row = rowb + i;
        if (row < nrows) {
            float v = acc[i] + bv;
            if (relu) v = fmaxf(v, 0.0f);
            out[(size_t)row * dout + ncol] = v;
        }
    }
}

// ---------------- log-softmax over 64 cols, one wave32 per row ----------------
__global__ void k_logsoftmax(float* __restrict__ out, int nrows, int dout) {
    int wave = threadIdx.x >> 5;
    int lane = threadIdx.x & 31;
    int row = blockIdx.x * (blockDim.x >> 5) + wave;
    if (row >= nrows) return;
    float* p = out + (size_t)row * dout;
    float v0 = p[lane];
    float v1 = p[lane + 32];
    float m = fmaxf(v0, v1);
    #pragma unroll
    for (int off = 16; off > 0; off >>= 1)
        m = fmaxf(m, __shfl_xor(m, off, 32));
    float s = __expf(v0 - m) + __expf(v1 - m);
    #pragma unroll
    for (int off = 16; off > 0; off >>= 1)
        s += __shfl_xor(s, off, 32);
    float lse = m + __logf(s);
    p[lane] = v0 - lse;
    p[lane + 32] = v1 - lse;
}

// ---------------- host launcher ----------------

extern "C" void kernel_launch(void* const* d_in, const int* in_sizes, int n_in,
                              void* d_out, int out_size, void* d_ws, size_t ws_size,
                              hipStream_t stream) {
    const float* x0  = (const float*)d_in[0];
    const int*   ei  = (const int*)d_in[1];
    const int N = in_sizes[0] / DH;
    const int E = in_sizes[1] / 2;
    const int* src = ei;
    const int* dst = ei + E;

    // workspace layout (256B aligned)
    char* ws = (char*)d_ws;
    size_t off = 0;
    auto take = [&](size_t bytes) {
        char* p = ws + off;
        off = (off + bytes + 255) & ~(size_t)255;
        return p;
    };
    float*  invdeg = (float*)take((size_t)N * 4);
    float*  h0     = (float*)take((size_t)N * DH * 4);
    float*  h1     = (float*)take((size_t)N * DH * 4);
    float*  agg    = (float*)take((size_t)N * DH * 4);
    __bf16* Abf    = (__bf16*)take((size_t)N * KTOT * 2);
    __bf16* Wbf    = (__bf16*)take((size_t)DH * KTOT * 2);

    const int T = 256;
    auto blocks = [](long long n, int t) { return (unsigned)((n + t - 1) / t); };

    // degree -> inv_deg (shared by all layers)
    k_zero<<<blocks(N, T), T, 0, stream>>>(invdeg, N);
    k_deg<<<blocks(E, T), T, 0, stream>>>(dst, invdeg, E);
    k_invdeg<<<blocks(N, T), T, 0, stream>>>(invdeg, N);

    const float* xin[4]  = { x0, h0, h1, h0 };
    float*       xout[4] = { h0, h1, h0, (float*)d_out };

    for (int l = 0; l < 4; ++l) {
        const float* Wl = (const float*)d_in[2 + 3 * l];
        const float* Wr = (const float*)d_in[3 + 3 * l];
        const float* b  = (const float*)d_in[4 + 3 * l];
        const int dout  = in_sizes[2 + 3 * l] / DH;   // 128,128,128,64
        const int coltiles = dout / 16;
        const int rowtiles = 8 / coltiles;

        k_zero<<<blocks((long long)N * DH, T), T, 0, stream>>>(agg, (long long)N * DH);
        k_scatter<<<blocks((long long)E * (DH / 4), T), T, 0, stream>>>(xin[l], src, dst, agg, E);
        k_prep<<<blocks((long long)N * (KTOT / 8), T), T, 0, stream>>>(xin[l], agg, invdeg, Abf, N);
        k_convw<<<blocks(dout * KTOT, T), T, 0, stream>>>(Wl, Wr, Wbf, dout);

        unsigned gx = (unsigned)((N + 16 * rowtiles - 1) / (16 * rowtiles));
        k_gemm<<<gx, T, 0, stream>>>(Abf, Wbf, b, xout[l], N, dout, rowtiles, coltiles,
                                     (l < 3) ? 1 : 0);
    }

    // final: log_softmax over 64 cols, in place in d_out
    k_logsoftmax<<<blocks((long long)N * 32, T), T, 0, stream>>>((float*)d_out, N, 64);
}
